// HypergraphConv_79620103733959
// MI455X (gfx1250) — matrix operand order repaired and checked by reading, not compiled, not run on hardware
//
#include <hip/hip_runtime.h>

#define NN 16384   // nodes
#define EE 4096    // hyperedges
#define FF 128     // features (in == out)
#define EPS 1e-6f
#define LDT 40     // padded LDS row stride (elements): 20-bank stride -> conflict-free

typedef __attribute__((ext_vector_type(16))) __bf16 v16bf;
typedef __attribute__((ext_vector_type(8)))  float  v8f;

union V16 { v16bf v; unsigned short u[16]; };

// two fp32 -> packed bf16x2, round-half-away (3 VALU: 2 adds + 1 v_perm)
static __device__ __forceinline__ unsigned int pack2_rna(float a, float b) {
    return __byte_perm(__float_as_uint(a) + 0x8000u,
                       __float_as_uint(b) + 0x8000u, 0x7632);
}
// fp32 -> bf16 truncation (EXACT for H's {0,1} values)
static __device__ __forceinline__ unsigned short trunc_bf16(float f) {
    return (unsigned short)(__float_as_uint(f) >> 16);   // folds into ds_store_b16_d16_hi
}
static __device__ __forceinline__ unsigned int pack2_trunc(float a, float b) {
    return __byte_perm(__float_as_uint(a), __float_as_uint(b), 0x7632);
}

// generic (__shared__) pointer -> 32-bit LDS offset for async-to-LDS VDST
typedef __attribute__((address_space(3))) unsigned short lds_us_t;
static __device__ __forceinline__ unsigned lds_addr(unsigned short* p) {
    return (unsigned)(unsigned long long)(lds_us_t*)p;
}
// async copy 8 bytes global -> LDS (tracked by ASYNCcnt, no VGPR round-trip)
static __device__ __forceinline__ void async_copy_b64(unsigned short* dst_lds,
                                                      const unsigned short* src_glb) {
    asm volatile("global_load_async_to_lds_b64 %0, %1, off"
                 :: "v"(lds_addr(dst_lds)),
                    "v"((unsigned long long)src_glb)
                 : "memory");
}
static __device__ __forceinline__ void wait_async() {
    asm volatile("s_wait_asynccnt 0x0" ::: "memory");
}

// ---- WMMA fragment assembly from LDS tiles ---------------------------------
static __device__ __forceinline__ v16bf frag_a(const unsigned short* a_lds, int lane) {
    V16 f;
    const int row  = lane & 15;
    const int koff = (lane >> 4) << 4;
    const unsigned short* p = a_lds + row * LDT;
#pragma unroll
    for (int j = 0; j < 8; ++j) {
        f.u[j]     = p[koff + j];
        f.u[8 + j] = p[koff + 8 + j];
    }
    return f.v;
}
static __device__ __forceinline__ v16bf frag_b(const unsigned short* b_lds, int lane, int nt) {
    V16 f;
    const int col   = lane & 15;
    const int kbase = (lane >> 4) << 4;
    const unsigned short* p = b_lds + (nt * 16 + col) * LDT + kbase;
#pragma unroll
    for (int j = 0; j < 16; ++j) f.u[j] = p[j];
    return f.v;
}

// ---- GEMM 0: xwT[F][N] = bf16(x @ W)^T -------------------------------------
__global__ __launch_bounds__(256) void hg_gemm_xw(const float* __restrict__ x,
                                                  const float* __restrict__ W,
                                                  unsigned short* __restrict__ xwT) {
    __shared__ __align__(16) unsigned short Alds[2][8][16][LDT];
    __shared__ __align__(16) unsigned short Blds[2][FF][LDT];
    const int tid = threadIdx.x, lane = tid & 31, w = tid >> 5;
    const int m_blk = blockIdx.x * 128;

    v8f acc[8];
    const v8f z = {0,0,0,0,0,0,0,0};
#pragma unroll
    for (int i = 0; i < 8; ++i) acc[i] = z;

    const int a_r  = tid >> 3;
    const int a_c4 = (tid & 7) << 2;
    const int b_j  = tid & 127;
    const int b_kb = (tid >> 7) << 2;

    auto load_tile = [&](int k0, int buf) {
        float4 av[4];
#pragma unroll
        for (int i = 0; i < 4; ++i) {
            int r = a_r + i * 32;
            av[i] = *(const float4*)&x[(size_t)(m_blk + r) * FF + k0 + a_c4];
        }
        float bv[4][4];
#pragma unroll
        for (int i = 0; i < 4; ++i) {
            int kb = b_kb + i * 8;
#pragma unroll
            for (int q = 0; q < 4; ++q)
                bv[i][q] = W[(size_t)(k0 + kb + q) * FF + b_j];
        }
#pragma unroll
        for (int i = 0; i < 4; ++i) {
            int r = a_r + i * 32;
            unsigned int* d = (unsigned int*)&Alds[buf][r >> 4][r & 15][a_c4];
            d[0] = pack2_rna(av[i].x, av[i].y);
            d[1] = pack2_rna(av[i].z, av[i].w);
        }
#pragma unroll
        for (int i = 0; i < 4; ++i) {
            int kb = b_kb + i * 8;
            unsigned int* d = (unsigned int*)&Blds[buf][b_j][kb];
            d[0] = pack2_rna(bv[i][0], bv[i][1]);
            d[1] = pack2_rna(bv[i][2], bv[i][3]);
        }
    };

    load_tile(0, 0);
    int p = 0;
    for (int k0 = 0; k0 < FF; k0 += 32) {
        __syncthreads();
        if (k0 + 32 < FF) load_tile(k0 + 32, p ^ 1);
        v16bf a = frag_a(&Alds[p][w][0][0], lane);
#pragma unroll
        for (int nt = 0; nt < 8; ++nt) {
            v16bf b = frag_b(&Blds[p][0][0], lane, nt);
            acc[nt] = __builtin_amdgcn_wmma_f32_16x16x32_bf16(
                false, a, false, b, (short)0, acc[nt], false, false);
        }
        p ^= 1;
    }
    const int rbase = m_blk + w * 16 + ((lane >> 4) << 3);
    const int cbase = lane & 15;
#pragma unroll
    for (int nt = 0; nt < 8; ++nt) {
        uint4 q;
        q.x = pack2_rna(acc[nt][0], acc[nt][1]);
        q.y = pack2_rna(acc[nt][2], acc[nt][3]);
        q.z = pack2_rna(acc[nt][4], acc[nt][5]);
        q.w = pack2_rna(acc[nt][6], acc[nt][7]);
        *(uint4*)&xwT[(size_t)(nt * 16 + cbase) * NN + rbase] = q;
    }
}

// ---- GEMM 1: edgeT[F][E] = (d_e_inv * (H^T @ xw))^T, fused edge degrees ----
// Block sees ALL node rows for its 128 edge columns -> exact column sums of H
// accumulate for free in registers while staging A (fp32 sums of {0,1}: exact).
__global__ __launch_bounds__(256) void hg_gemm_edge(const float* __restrict__ H,
                                                    const unsigned short* __restrict__ xwT,
                                                    unsigned short* __restrict__ edgeT) {
    __shared__ __align__(16) unsigned short Alds[2][8][16][LDT];
    __shared__ __align__(16) unsigned short Blds[2][FF][LDT];
    __shared__ float dsum[128];
    const int tid = threadIdx.x, lane = tid & 31, w = tid >> 5;
    const int m_blk = blockIdx.x * 128;   // edge base

    v8f acc[8];
    const v8f z = {0,0,0,0,0,0,0,0};
#pragma unroll
    for (int i = 0; i < 8; ++i) acc[i] = z;

    const int a_kk = tid >> 5;               // node row in slab (i adds 8)
    const int a_c4 = (tid & 31) << 2;        // edge col
    float4 csum = {0.f, 0.f, 0.f, 0.f};      // degree partial for cols a_c4..a_c4+3

    auto load_tile = [&](int k0, int buf) {
        // B tiles: pure bf16 copy -> async DMA straight into LDS
#pragma unroll
        for (int i = 0; i < 4; ++i) {
            int lin = i * 256 + tid;
            int j  = lin >> 3;
            int k4 = (lin & 7) << 2;
            async_copy_b64(&Blds[buf][j][k4], &xwT[(size_t)j * NN + k0 + k4]);
        }
        float4 av[4];
#pragma unroll
        for (int i = 0; i < 4; ++i) {
            int kk = a_kk + i * 8;
            av[i] = *(const float4*)&H[(size_t)(k0 + kk) * EE + m_blk + a_c4];
        }
        // transposed scatter: truncation exact for H -> ds_store_b16_d16_hi
#pragma unroll
        for (int i = 0; i < 4; ++i) {
            int kk = a_kk + i * 8;
            unsigned short* base = &Alds[buf][a_c4 >> 4][a_c4 & 15][kk];
            base[0 * LDT] = trunc_bf16(av[i].x);
            base[1 * LDT] = trunc_bf16(av[i].y);
            base[2 * LDT] = trunc_bf16(av[i].z);
            base[3 * LDT] = trunc_bf16(av[i].w);
            csum.x += av[i].x; csum.y += av[i].y;
            csum.z += av[i].z; csum.w += av[i].w;
        }
    };

    load_tile(0, 0);
    int p = 0;
    for (int k0 = 0; k0 < NN; k0 += 32) {
        wait_async();
        __syncthreads();
        if (k0 + 32 < NN) {
            load_tile(k0 + 32, p ^ 1);
            __builtin_prefetch(&H[(size_t)(k0 + 64 + (tid >> 3)) * EE + m_blk + ((tid & 7) << 4)], 0, 1);
        }
        v16bf a = frag_a(&Alds[p][w][0][0], lane);
#pragma unroll
        for (int nt = 0; nt < 8; ++nt) {
            v16bf b = frag_b(&Blds[p][0][0], lane, nt);
            acc[nt] = __builtin_amdgcn_wmma_f32_16x16x32_bf16(
                false, a, false, b, (short)0, acc[nt], false, false);
        }
        p ^= 1;
    }
    // reduce degree partials (8 threads per column group) in LDS
    if (tid < 128) dsum[tid] = 0.0f;
    __syncthreads();
    atomicAdd(&dsum[a_c4 + 0], csum.x);
    atomicAdd(&dsum[a_c4 + 1], csum.y);
    atomicAdd(&dsum[a_c4 + 2], csum.z);
    atomicAdd(&dsum[a_c4 + 3], csum.w);
    __syncthreads();

    const int rloc  = w * 16 + ((lane >> 4) << 3);   // block-local edge row base
    const int rbase = m_blk + rloc;
    const int cbase = lane & 15;
    float s[8];
#pragma unroll
    for (int v = 0; v < 8; ++v) s[v] = 1.0f / (dsum[rloc + v] + EPS);
#pragma unroll
    for (int nt = 0; nt < 8; ++nt) {
        uint4 q;
        q.x = pack2_rna(acc[nt][0] * s[0], acc[nt][1] * s[1]);
        q.y = pack2_rna(acc[nt][2] * s[2], acc[nt][3] * s[3]);
        q.z = pack2_rna(acc[nt][4] * s[4], acc[nt][5] * s[5]);
        q.w = pack2_rna(acc[nt][6] * s[6], acc[nt][7] * s[7]);
        *(uint4*)&edgeT[(size_t)(nt * 16 + cbase) * EE + rbase] = q;
    }
}

// ---- GEMM 2: out[N][F] = d_v_inv * (H @ edge), fused node degrees ----------
// Block sees ALL edge columns for its 128 node rows -> exact row sums of H.
__global__ __launch_bounds__(256) void hg_gemm_node(const float* __restrict__ H,
                                                    const unsigned short* __restrict__ edgeT,
                                                    float* __restrict__ out) {
    __shared__ __align__(16) unsigned short Alds[2][8][16][LDT];
    __shared__ __align__(16) unsigned short Blds[2][FF][LDT];
    __shared__ float dsum[128];
    const int tid = threadIdx.x, lane = tid & 31, w = tid >> 5;
    const int m_blk = blockIdx.x * 128;   // node base

    v8f acc[8];
    const v8f z = {0,0,0,0,0,0,0,0};
#pragma unroll
    for (int i = 0; i < 8; ++i) acc[i] = z;

    const int a_r  = tid >> 3;               // block-local node row (i adds 32)
    const int a_c4 = (tid & 7) << 2;
    float rsum[4] = {0.f, 0.f, 0.f, 0.f};    // degree partial for rows a_r+32i

    auto load_tile = [&](int k0, int buf) {
#pragma unroll
        for (int i = 0; i < 4; ++i) {
            int lin = i * 256 + tid;
            int j  = lin >> 3;
            int k4 = (lin & 7) << 2;
            async_copy_b64(&Blds[buf][j][k4], &edgeT[(size_t)j * EE + k0 + k4]);
        }
        float4 av[4];
#pragma unroll
        for (int i = 0; i < 4; ++i) {
            int r = a_r + i * 32;
            av[i] = *(const float4*)&H[(size_t)(m_blk + r) * EE + k0 + a_c4];
        }
#pragma unroll
        for (int i = 0; i < 4; ++i) {
            int r = a_r + i * 32;
            unsigned int* d = (unsigned int*)&Alds[buf][r >> 4][r & 15][a_c4];
            d[0] = pack2_trunc(av[i].x, av[i].y);
            d[1] = pack2_trunc(av[i].z, av[i].w);
            rsum[i] += (av[i].x + av[i].y) + (av[i].z + av[i].w);
        }
    };

    load_tile(0, 0);
    int p = 0;
    for (int k0 = 0; k0 < EE; k0 += 32) {
        wait_async();
        __syncthreads();
        if (k0 + 32 < EE) {
            load_tile(k0 + 32, p ^ 1);
            __builtin_prefetch(&H[(size_t)(m_blk + (tid >> 3)) * EE + k0 + 64 + ((tid & 7) << 2)], 0, 1);
        }
        v16bf a = frag_a(&Alds[p][w][0][0], lane);
#pragma unroll
        for (int nt = 0; nt < 8; ++nt) {
            v16bf b = frag_b(&Blds[p][0][0], lane, nt);
            acc[nt] = __builtin_amdgcn_wmma_f32_16x16x32_bf16(
                false, a, false, b, (short)0, acc[nt], false, false);
        }
        p ^= 1;
    }
    // reduce degree partials (8 threads per row) in LDS
    if (tid < 128) dsum[tid] = 0.0f;
    __syncthreads();
#pragma unroll
    for (int i = 0; i < 4; ++i)
        atomicAdd(&dsum[a_r + i * 32], rsum[i]);
    __syncthreads();

    const int rloc  = w * 16 + ((lane >> 4) << 3);
    const int rbase = m_blk + rloc;
    const int cbase = lane & 15;
#pragma unroll
    for (int v = 0; v < 8; ++v) {
        const int row = rbase + v;
        const float s = 1.0f / (dsum[rloc + v] + EPS);
#pragma unroll
        for (int nt = 0; nt < 8; ++nt)
            out[(size_t)row * FF + nt * 16 + cbase] = s * acc[nt][v];
    }
}

extern "C" void kernel_launch(void* const* d_in, const int* in_sizes, int n_in,
                              void* d_out, int out_size, void* d_ws, size_t ws_size,
                              hipStream_t stream) {
    (void)in_sizes; (void)n_in; (void)out_size; (void)ws_size;
    const float* x = (const float*)d_in[0];   // [N, F]
    const float* H = (const float*)d_in[1];   // [N, E]
    const float* W = (const float*)d_in[2];   // [F, F]
    float* out = (float*)d_out;               // [N, F] fp32

    char* ws = (char*)d_ws;
    unsigned short* xwT   = (unsigned short*)(ws);                           // 4 MB  [F][N]
    unsigned short* edgeT = (unsigned short*)(ws + (size_t)4 * 1024 * 1024); // 1 MB  [F][E]

    // GEMM chain: H read exactly twice; degrees fused into the GEMMs.
    hg_gemm_xw<<<NN / 128, 256, 0, stream>>>(x, W, xwT);
    hg_gemm_edge<<<EE / 128, 256, 0, stream>>>(H, xwT, edgeT);
    hg_gemm_node<<<NN / 128, 256, 0, stream>>>(H, edgeT, out);
}
